// OptimizedMOE_65180423685432
// MI455X (gfx1250) — matrix operand order repaired and checked by reading, not compiled, not run on hardware
//
#include <hip/hip_runtime.h>
#include <hip/hip_bf16.h>

#define EPSV   1e-5f
#define NE     4
#define B_     16
#define C_     256
#define O_     256
#define HID_   512
#define HW_    4096   // 64*64
#define MT     64     // pixels per workgroup tile
#define CPAD   264    // padded K stride for x tile (f16 elems)
#define HPAD   520    // padded K stride for h1 tile (f16 elems)

typedef __attribute__((ext_vector_type(16))) _Float16 v16h;
typedef __attribute__((ext_vector_type(8)))  _Float16 v8h;
typedef __attribute__((ext_vector_type(8)))  float    v8f;

union F16x16 { v16h v; v8h h[2]; };

// ---------------- kernel 1: global average pool  pooled[b,c] = mean_hw x ----
__global__ __launch_bounds__(256) void k_pool(const float* __restrict__ x,
                                              float* __restrict__ pooled) {
  __shared__ float red[256];
  int bc = blockIdx.x;                       // b*C + c
  const float* p = x + (size_t)bc * HW_;
  float s = 0.f;
  for (int i = threadIdx.x; i < HW_; i += 256) s += p[i];
  red[threadIdx.x] = s;
  __syncthreads();
  for (int off = 128; off > 0; off >>= 1) {
    if ((int)threadIdx.x < off) red[threadIdx.x] += red[threadIdx.x + off];
    __syncthreads();
  }
  if (threadIdx.x == 0) pooled[bc] = red[0] * (1.0f / HW_);
}

// ---------------- kernel 2: router logits -> softmax -> top2 renorm ---------
__global__ __launch_bounds__(64) void k_router(const float* __restrict__ pooled,
                                               const float* __restrict__ rw,
                                               const float* __restrict__ rb,
                                               int* __restrict__ sel_id,
                                               float* __restrict__ sel_w) {
  __shared__ float lg[B_][NE];
  int t = threadIdx.x;
  if (t < B_ * NE) {
    int b = t / NE, e = t % NE;
    const float* pv = pooled + b * C_;
    const float* wv = rw + e * C_;
    float s = rb[e];
    for (int c = 0; c < C_; ++c) s += pv[c] * wv[c];
    lg[b][e] = s;
  }
  __syncthreads();
  if (t < B_) {
    float m = lg[t][0];
    for (int e = 1; e < NE; ++e) m = fmaxf(m, lg[t][e]);
    float p[NE], sum = 0.f;
    for (int e = 0; e < NE; ++e) { p[e] = __expf(lg[t][e] - m); sum += p[e]; }
    float inv = 1.f / sum;
    for (int e = 0; e < NE; ++e) p[e] *= inv;
    int i0 = 0;
    for (int e = 1; e < NE; ++e) if (p[e] > p[i0]) i0 = e;
    int i1 = -1;
    for (int e = 0; e < NE; ++e) { if (e == i0) continue; if (i1 < 0 || p[e] > p[i1]) i1 = e; }
    float ws2 = p[i0] + p[i1];
    sel_id[t * 2 + 0] = i0;  sel_id[t * 2 + 1] = i1;
    sel_w[t * 2 + 0] = p[i0] / ws2;  sel_w[t * 2 + 1] = p[i1] / ws2;
  }
}

// ---------------- kernel 3: fold BN scale into weights, convert to f16 ------
__global__ __launch_bounds__(256) void k_prep(
    const float* __restrict__ sw, const float* __restrict__ sg,
    const float* __restrict__ sv, const float* __restrict__ sm,
    const float* __restrict__ sbeta,
    const float* __restrict__ w1, const float* __restrict__ g1,
    const float* __restrict__ v1, const float* __restrict__ m1,
    const float* __restrict__ b1,
    const float* __restrict__ w2, const float* __restrict__ g2,
    const float* __restrict__ v2, const float* __restrict__ m2,
    const float* __restrict__ b2,
    _Float16* __restrict__ swh, _Float16* __restrict__ w1h,
    _Float16* __restrict__ w2h,
    float* __restrict__ sbias, float* __restrict__ bias1,
    float* __restrict__ bias2) {
  const int N1 = NE * HID_ * C_;
  const int N2 = NE * O_ * HID_;
  const int N3 = O_ * C_;
  int i = blockIdx.x * blockDim.x + threadIdx.x;
  if (i < N1) {
    int ed = i / C_;
    float sc = g1[ed] * rsqrtf(v1[ed] + EPSV);
    w1h[i] = (_Float16)(w1[i] * sc);
  } else if (i < N1 + N2) {
    int j = i - N1;
    int eo = j / HID_;
    float sc = g2[eo] * rsqrtf(v2[eo] + EPSV);
    w2h[j] = (_Float16)(w2[j] * sc);
  } else if (i < N1 + N2 + N3) {
    int j = i - N1 - N2;
    int o = j / C_;
    float sc = sg[o] * rsqrtf(sv[o] + EPSV);
    swh[j] = (_Float16)(sw[j] * sc);
  } else {
    int j = i - N1 - N2 - N3;
    if (j < NE * HID_) {
      float sc = g1[j] * rsqrtf(v1[j] + EPSV);
      bias1[j] = b1[j] - m1[j] * sc;
    } else if (j < NE * HID_ + NE * O_) {
      int k = j - NE * HID_;
      float sc = g2[k] * rsqrtf(v2[k] + EPSV);
      bias2[k] = b2[k] - m2[k] * sc;
    } else if (j < NE * HID_ + NE * O_ + O_) {
      int k = j - NE * HID_ - NE * O_;
      float sc = sg[k] * rsqrtf(sv[k] + EPSV);
      sbias[k] = sbeta[k] - sm[k] * sc;
    }
  }
}

__device__ __forceinline__ float silu_f(float v) {
  return v / (1.f + __expf(-v));
}

__device__ __forceinline__ void load_ax(const _Float16* __restrict__ xrow,
                                        int hiK, F16x16* ax) {
  #pragma unroll
  for (int k = 0; k < 8; ++k) {
    ax[k].h[0] = *(const v8h*)(xrow + k * 32 + hiK * 16);
    ax[k].h[1] = *(const v8h*)(xrow + k * 32 + hiK * 16 + 8);
  }
}

// ---------------- kernel 4: main WMMA pipeline ------------------------------
// Block = 256 threads = 8 waves; covers 64 pixels of one image.
// waves: strip = wid&3 (16-pixel M strip), nh = wid>>2 (N half).
// All GEMM loops process N-tile PAIRS with dual accumulators: A fragments are
// reused across two WMMA chains (halves LDS A traffic in GEMM2, doubles XDL ILP).
__global__ __launch_bounds__(256) void k_main(
    const float* __restrict__ x,
    const _Float16* __restrict__ swh, const _Float16* __restrict__ w1h,
    const _Float16* __restrict__ w2h,
    const float* __restrict__ sbias, const float* __restrict__ bias1,
    const float* __restrict__ bias2,
    const int* __restrict__ sel_id, const float* __restrict__ sel_w,
    float* __restrict__ out) {
  extern __shared__ char smem[];
  _Float16* xl  = (_Float16*)smem;                         // [MT][CPAD]
  _Float16* h1l = (_Float16*)(smem + MT * CPAD * 2);       // [4][16][HPAD]

  const int b    = blockIdx.x / (HW_ / MT);
  const int tile = blockIdx.x % (HW_ / MT);
  const int p0   = tile * MT;
  const float* xb = x + (size_t)b * C_ * HW_ + p0;

  // ---- stage x tile -> LDS as f16, [m][k] layout (float4 global loads) ----
  {
    int m4 = (threadIdx.x & 15) * 4;
    int c0 = threadIdx.x >> 4;           // 0..15
    for (int c = c0; c < C_; c += 16) {
      float4 v = *(const float4*)(xb + (size_t)c * HW_ + m4);
      xl[(m4 + 0) * CPAD + c] = (_Float16)v.x;
      xl[(m4 + 1) * CPAD + c] = (_Float16)v.y;
      xl[(m4 + 2) * CPAD + c] = (_Float16)v.z;
      xl[(m4 + 3) * CPAD + c] = (_Float16)v.w;
    }
  }
  __syncthreads();

  const int wid   = threadIdx.x >> 5;
  const int lane  = threadIdx.x & 31;
  const int strip = wid & 3;
  const int nh    = wid >> 2;
  const int l16   = lane & 15;
  const int hiK   = (lane >> 4) & 1;

  const _Float16* xrow = xl + (strip * 16 + l16) * CPAD;

  const int e0 = sel_id[b * 2 + 0], e1 = sel_id[b * 2 + 1];
  const float cwA = sel_w[b * 2 + 0], cwB = sel_w[b * 2 + 1];

  float* outb = out + (size_t)b * O_ * HW_ + p0;
  const int mbase = strip * 16 + hiK * 8;      // pixel row base for C/D tiles

  F16x16 ax[8];

  // ---- shared expert: O=256, this wave does 8 of 16 N tiles (4 pairs) ----
  load_ax(xrow, hiK, ax);
  for (int np = 0; np < 4; ++np) {
    const int n0 = (nh * 8 + np * 2) * 16;
    v8f accA = {}, accB = {};
    const _Float16* wbA = swh + (size_t)(n0 + l16) * C_ + hiK * 16;
    const _Float16* wbB = wbA + (size_t)16 * C_;
    __builtin_prefetch(wbA + (size_t)32 * C_, 0, 0);
    #pragma unroll
    for (int k = 0; k < 8; ++k) {
      v16h bfA = *(const v16h*)(wbA + k * 32);
      v16h bfB = *(const v16h*)(wbB + k * 32);
      accA = __builtin_amdgcn_wmma_f32_16x16x32_f16(false, ax[k].v, false, bfA,
                                                    (short)0, accA, false, false);
      accB = __builtin_amdgcn_wmma_f32_16x16x32_f16(false, ax[k].v, false, bfB,
                                                    (short)0, accB, false, false);
    }
    const float bvA = sbias[n0 + l16];
    const float bvB = sbias[n0 + 16 + l16];
    float rA[8], rB[8];
    #pragma unroll
    for (int i = 0; i < 8; ++i) {
      rA[i] = silu_f(accA[i] + bvA);
      rB[i] = silu_f(accB[i] + bvB);
    }
    float* opA = outb + (size_t)(n0 + l16) * HW_ + mbase;
    float* opB = opA + (size_t)16 * HW_;
    *(float4*)(opA)     = make_float4(rA[0], rA[1], rA[2], rA[3]);
    *(float4*)(opA + 4) = make_float4(rA[4], rA[5], rA[6], rA[7]);
    *(float4*)(opB)     = make_float4(rB[0], rB[1], rB[2], rB[3]);
    *(float4*)(opB + 4) = make_float4(rB[4], rB[5], rB[6], rB[7]);
  }

  // ---- two selected experts ----
  _Float16* h1s = h1l + strip * 16 * HPAD;     // this strip's h1 tile [16][HPAD]
  for (int kk = 0; kk < 2; ++kk) {
    const int   e  = kk ? e1 : e0;
    const float cw = kk ? cwB : cwA;
    const _Float16* w1e = w1h + (size_t)e * HID_ * C_;
    const _Float16* w2e = w2h + (size_t)e * O_ * HID_;
    const float* b1e = bias1 + e * HID_;
    const float* b2e = bias2 + e * O_;

    // GEMM1: h1[16, 512] = x @ w1'^T ; 16 tiles per wave = 8 pairs
    load_ax(xrow, hiK, ax);
    for (int np = 0; np < 8; ++np) {
      const int n0 = (nh * 16 + np * 2) * 16;
      v8f accA = {}, accB = {};
      const _Float16* wbA = w1e + (size_t)(n0 + l16) * C_ + hiK * 16;
      const _Float16* wbB = wbA + (size_t)16 * C_;
      __builtin_prefetch(wbA + (size_t)32 * C_, 0, 0);
      #pragma unroll
      for (int k = 0; k < 8; ++k) {
        v16h bfA = *(const v16h*)(wbA + k * 32);
        v16h bfB = *(const v16h*)(wbB + k * 32);
        accA = __builtin_amdgcn_wmma_f32_16x16x32_f16(false, ax[k].v, false, bfA,
                                                      (short)0, accA, false, false);
        accB = __builtin_amdgcn_wmma_f32_16x16x32_f16(false, ax[k].v, false, bfB,
                                                      (short)0, accB, false, false);
      }
      const float bvA = b1e[n0 + l16];
      const float bvB = b1e[n0 + 16 + l16];
      const int mr = hiK * 8;
      #pragma unroll
      for (int i = 0; i < 8; ++i) {
        float vA = silu_f(accA[i] + bvA);
        float vB = silu_f(accB[i] + bvB);
        h1s[(mr + i) * HPAD + n0 + l16]      = (_Float16)vA;  // D layout -> [m][d]
        h1s[(mr + i) * HPAD + n0 + 16 + l16] = (_Float16)vB;
      }
    }
    __syncthreads();   // both N-half waves finished writing this strip's h1

    // GEMM2: h2[16, 256] = h1 @ w2'^T ; 8 tiles per wave = 4 pairs.
    // A fragment (LDS) shared by both accumulators of the pair.
    const _Float16* arow = h1s + l16 * HPAD;
    for (int np = 0; np < 4; ++np) {
      const int n0 = (nh * 8 + np * 2) * 16;
      v8f accA = {}, accB = {};
      const _Float16* wbA = w2e + (size_t)(n0 + l16) * HID_ + hiK * 16;
      const _Float16* wbB = wbA + (size_t)16 * HID_;
      #pragma unroll 4
      for (int k = 0; k < 16; ++k) {
        F16x16 a2;
        a2.h[0] = *(const v8h*)(arow + k * 32 + hiK * 16);
        a2.h[1] = *(const v8h*)(arow + k * 32 + hiK * 16 + 8);
        v16h bfA = *(const v16h*)(wbA + k * 32);
        v16h bfB = *(const v16h*)(wbB + k * 32);
        accA = __builtin_amdgcn_wmma_f32_16x16x32_f16(false, a2.v, false, bfA,
                                                      (short)0, accA, false, false);
        accB = __builtin_amdgcn_wmma_f32_16x16x32_f16(false, a2.v, false, bfB,
                                                      (short)0, accB, false, false);
      }
      const float bvA = b2e[n0 + l16];
      const float bvB = b2e[n0 + 16 + l16];
      float* opA = outb + (size_t)(n0 + l16) * HW_ + mbase;
      float* opB = opA + (size_t)16 * HW_;
      float4 a0 = *(float4*)(opA);
      float4 a1 = *(float4*)(opA + 4);
      float4 b0 = *(float4*)(opB);
      float4 b1v = *(float4*)(opB + 4);
      a0.x += (accA[0] + bvA) * cw;  a0.y += (accA[1] + bvA) * cw;
      a0.z += (accA[2] + bvA) * cw;  a0.w += (accA[3] + bvA) * cw;
      a1.x += (accA[4] + bvA) * cw;  a1.y += (accA[5] + bvA) * cw;
      a1.z += (accA[6] + bvA) * cw;  a1.w += (accA[7] + bvA) * cw;
      b0.x += (accB[0] + bvB) * cw;  b0.y += (accB[1] + bvB) * cw;
      b0.z += (accB[2] + bvB) * cw;  b0.w += (accB[3] + bvB) * cw;
      b1v.x += (accB[4] + bvB) * cw; b1v.y += (accB[5] + bvB) * cw;
      b1v.z += (accB[6] + bvB) * cw; b1v.w += (accB[7] + bvB) * cw;
      *(float4*)(opA)     = a0;
      *(float4*)(opA + 4) = a1;
      *(float4*)(opB)     = b0;
      *(float4*)(opB + 4) = b1v;
    }
    __syncthreads();   // h1 consumed; safe to overwrite for next expert
  }
}

// ---------------- host side -------------------------------------------------
extern "C" void kernel_launch(void* const* d_in, const int* in_sizes, int n_in,
                              void* d_out, int out_size, void* d_ws, size_t ws_size,
                              hipStream_t stream) {
  const float* x        = (const float*)d_in[0];
  const float* router_w = (const float*)d_in[1];
  const float* router_b = (const float*)d_in[2];
  const float* shared_w = (const float*)d_in[3];
  const float* shared_g = (const float*)d_in[4];
  const float* shared_bt= (const float*)d_in[5];
  const float* shared_m = (const float*)d_in[6];
  const float* shared_v = (const float*)d_in[7];
  const float* w1 = (const float*)d_in[8];
  const float* g1 = (const float*)d_in[9];
  const float* b1 = (const float*)d_in[10];
  const float* m1 = (const float*)d_in[11];
  const float* v1 = (const float*)d_in[12];
  const float* w2 = (const float*)d_in[13];
  const float* g2 = (const float*)d_in[14];
  const float* b2 = (const float*)d_in[15];
  const float* m2 = (const float*)d_in[16];
  const float* v2 = (const float*)d_in[17];
  float* out = (float*)d_out;

  char* ws = (char*)d_ws;
  size_t off = 0;
  auto carve = [&](size_t bytes) -> char* {
    char* p = ws + off;
    off = (off + bytes + 255) & ~(size_t)255;
    return p;
  };
  _Float16* w1h   = (_Float16*)carve((size_t)NE * HID_ * C_ * 2);
  _Float16* w2h   = (_Float16*)carve((size_t)NE * O_ * HID_ * 2);
  _Float16* swh   = (_Float16*)carve((size_t)O_ * C_ * 2);
  float*    bias1 = (float*)carve((size_t)NE * HID_ * 4);
  float*    bias2 = (float*)carve((size_t)NE * O_ * 4);
  float*    sbias = (float*)carve((size_t)O_ * 4);
  float*    pooled= (float*)carve((size_t)B_ * C_ * 4);
  int*      sel_id= (int*)carve((size_t)B_ * 2 * 4);
  float*    sel_w = (float*)carve((size_t)B_ * 2 * 4);

  k_pool<<<B_ * C_, 256, 0, stream>>>(x, pooled);
  k_router<<<1, 64, 0, stream>>>(pooled, router_w, router_b, sel_id, sel_w);

  const int NTOT = NE * HID_ * C_ + NE * O_ * HID_ + O_ * C_
                 + NE * HID_ + NE * O_ + O_;
  k_prep<<<(NTOT + 255) / 256, 256, 0, stream>>>(
      shared_w, shared_g, shared_v, shared_m, shared_bt,
      w1, g1, v1, m1, b1, w2, g2, v2, m2, b2,
      swh, w1h, w2h, sbias, bias1, bias2);

  const int smemBytes = MT * CPAD * 2 + 4 * 16 * HPAD * 2;   // ~100 KB
  (void)hipFuncSetAttribute((const void*)k_main,
                            hipFuncAttributeMaxDynamicSharedMemorySize,
                            smemBytes);
  k_main<<<B_ * (HW_ / MT), 256, smemBytes, stream>>>(
      x, swh, w1h, w2h, sbias, bias1, bias2, sel_id, sel_w, out);
}
